// AbsolutePE_LM_41661182771530
// MI455X (gfx1250) — compile-verified, matrix operand order checked by reading, not run on hardware
//
#include <hip/hip_runtime.h>
#include <math.h>

// ---------------------------------------------------------------------------
// Problem constants (AbsolutePE LM): V=32000, D=1024, MAXLEN=2048, B=4, C=2048
// ---------------------------------------------------------------------------
#define DIM   1024
#define CTX   2048
#define BATCH 4
#define VOCAB 32000
#define MROWS (BATCH * CTX)   // 8192

typedef __attribute__((ext_vector_type(16))) __bf16        v16bf;
typedef __attribute__((ext_vector_type(8)))  float         v8f;
typedef __attribute__((ext_vector_type(4)))  unsigned int  u32x4;

union Frag {
    v16bf          v;
    u32x4          u[2];
    unsigned short s[16];
};

static __device__ __forceinline__ unsigned short f2bf(float f) {
    unsigned int u = __builtin_bit_cast(unsigned int, f);
    u = (u + 0x7FFFu + ((u >> 16) & 1u)) >> 16;   // round-to-nearest-even
    return (unsigned short)u;
}

// Load one 16-element bf16 fragment: two contiguous 16B chunks at p and p+16
// (per-lane chunk base already includes the +8 offset for lanes 16-31, matching
// the CDNA5 16-bit A/B fragment layout: K={0..7,16..23} / K={8..15,24..31}).
static __device__ __forceinline__ v16bf ldfrag(const unsigned short* p) {
    Frag f;
    f.u[0] = *(const u32x4*)(p);
    f.u[1] = *(const u32x4*)(p + 16);
    return f.v;
}

static __device__ __forceinline__ v8f wmma_bf16(v16bf a, v16bf b, v8f c) {
    return __builtin_amdgcn_wmma_f32_16x16x32_bf16(false, a, false, b,
                                                   (short)0, c, false, false);
}

// Async global->LDS 16B copy (CDNA5 GLOBAL_LOAD_ASYNC_TO_LDS_B128, GV mode).
// VDST = per-lane LDS byte offset (low 32 bits of the generic pointer),
// VADDR = per-lane 64-bit global address. Tracked with ASYNCcnt.
static __device__ __forceinline__ void async_ld16(const void* lds_ptr,
                                                  const unsigned short* gptr) {
    unsigned loff = (unsigned)(unsigned long long)(uintptr_t)lds_ptr;
    asm volatile("global_load_async_to_lds_b128 %0, %1, off"
                 :: "v"(loff), "v"(gptr) : "memory");
}

static __device__ __forceinline__ void wait_async0() {
    asm volatile("s_wait_asynccnt 0x0" ::: "memory");
}

// ---------------------------------------------------------------------------
// Kernel 1: f32 -> bf16 downconvert
// ---------------------------------------------------------------------------
__global__ void k_cvt_bf16(const float* __restrict__ src,
                           unsigned short* __restrict__ dst, int n) {
    int i = blockIdx.x * 256 + threadIdx.x;
    if (i < n) dst[i] = f2bf(src[i]);
}

// ---------------------------------------------------------------------------
// Kernel 2: h = embed[x] + pe  (f32 for residual, bf16 for matmuls)
// ---------------------------------------------------------------------------
__global__ void k_embed(const int* __restrict__ x,
                        const float* __restrict__ embed,
                        const float* __restrict__ pe,
                        float* __restrict__ hF,
                        unsigned short* __restrict__ hB) {
    int i = blockIdx.x * 256 + threadIdx.x;          // over MROWS*DIM
    int bc = i >> 10;                                 // /DIM
    int d  = i & (DIM - 1);
    int c  = bc & (CTX - 1);
    int tok = x[bc];
    float v = embed[tok * DIM + d] + pe[c * DIM + d];
    hF[i] = v;
    hB[i] = f2bf(v);
}

// ---------------------------------------------------------------------------
// Kernel 3: Q/K/V = h @ W^T   (M=8192, N=1024, K=1024), bf16 in/out, f32 acc.
// Each wave: 32x64 tile, 8 accumulators, K stepped by 32.
// ---------------------------------------------------------------------------
__global__ __launch_bounds__(256)
void k_gemm_qkv(const unsigned short* __restrict__ hB,
                const unsigned short* __restrict__ WqB,
                const unsigned short* __restrict__ WkB,
                const unsigned short* __restrict__ WvB,
                unsigned short* __restrict__ Qb,
                unsigned short* __restrict__ Kb,
                unsigned short* __restrict__ Vb) {
    const unsigned short* W = (blockIdx.z == 0) ? WqB : (blockIdx.z == 1) ? WkB : WvB;
    unsigned short*       O = (blockIdx.z == 0) ? Qb  : (blockIdx.z == 1) ? Kb  : Vb;

    int wid = threadIdx.x >> 5, lane = threadIdx.x & 31;
    int hi8 = (lane >> 4) * 8, lm = lane & 15;
    int g   = blockIdx.x * 8 + wid;          // 4096 wave tiles
    int m0  = (g >> 4) * 32;                 // 256 m-tiles of 32
    int n0  = (g & 15) * 64;                 // 16  n-tiles of 64

    const unsigned short* A0 = hB + (m0 + lm) * DIM + hi8;
    const unsigned short* A1 = A0 + 16 * DIM;
    const unsigned short* B0 = W + (n0 + lm) * DIM + hi8;

    v8f z = {};
    v8f acc[2][4];
    #pragma unroll
    for (int mi = 0; mi < 2; ++mi)
        #pragma unroll
        for (int nt = 0; nt < 4; ++nt) acc[mi][nt] = z;

    for (int k0 = 0; k0 < DIM; k0 += 32) {
        v16bf a0 = ldfrag(A0 + k0);
        v16bf a1 = ldfrag(A1 + k0);
        #pragma unroll
        for (int nt = 0; nt < 4; ++nt) {
            v16bf b = ldfrag(B0 + nt * 16 * DIM + k0);
            acc[0][nt] = wmma_bf16(a0, b, acc[0][nt]);
            acc[1][nt] = wmma_bf16(a1, b, acc[1][nt]);
        }
    }

    #pragma unroll
    for (int mi = 0; mi < 2; ++mi)
        #pragma unroll
        for (int nt = 0; nt < 4; ++nt)
            #pragma unroll
            for (int r = 0; r < 8; ++r) {
                int row = m0 + mi * 16 + r + hi8;
                int col = n0 + nt * 16 + lm;
                O[row * DIM + col] = f2bf(acc[mi][nt][r]);
            }
}

// ---------------------------------------------------------------------------
// Kernel 4: causal flash attention + residual.
// Grid (C/16, B), block 512 (16 waves). One 16-row query block per workgroup;
// wave w owns output D-slice [w*64, w*64+64). Key blocks of 32.
// ---------------------------------------------------------------------------
__global__ __launch_bounds__(512)
void k_attn(const unsigned short* __restrict__ Qb,
            const unsigned short* __restrict__ Kb,
            const unsigned short* __restrict__ Vb,
            const float* __restrict__ hF,
            unsigned short* __restrict__ hidB) {
    __shared__ alignas(16) float          S[16 * 32];      // score tile
    __shared__ alignas(16) unsigned short P[16 * 32];      // probs (bf16)
    __shared__ float mrow[16], lrow[16], crow[16];

    int tid = threadIdx.x, wid = tid >> 5, lane = tid & 31;
    int hi8 = (lane >> 4) * 8, lm = lane & 15;
    int b  = blockIdx.y;
    int q0 = blockIdx.x * 16;                 // query row within batch
    int bq = b * CTX + q0;

    if (tid < 16) { mrow[tid] = -INFINITY; lrow[tid] = 0.0f; }

    // Preload Q fragments for this wave's D slice (K = wid*64 .. +64)
    const unsigned short* Qr = Qb + (bq + lm) * DIM + wid * 64 + hi8;
    v16bf aq0 = ldfrag(Qr);
    v16bf aq1 = ldfrag(Qr + 32);

    v8f z = {};
    v8f acc[4] = {z, z, z, z};

    int nkb = (q0 + 15) / 32 + 1;
    for (int kb = 0; kb < nkb; ++kb) {
        int k0g = kb * 32;
        S[tid] = 0.0f;                         // blockDim == 512 == tile size
        __syncthreads();

        // ---- partial S = Q * K^T over this wave's D slice, reduce via LDS
        #pragma unroll
        for (int nt = 0; nt < 2; ++nt) {
            const unsigned short* Kr =
                Kb + (b * CTX + k0g + nt * 16 + lm) * DIM + wid * 64 + hi8;
            v8f s = z;
            s = wmma_bf16(aq0, ldfrag(Kr),      s);
            s = wmma_bf16(aq1, ldfrag(Kr + 32), s);
            #pragma unroll
            for (int r = 0; r < 8; ++r)
                atomicAdd(&S[(r + hi8) * 32 + nt * 16 + lm], s[r]);
        }
        __syncthreads();

        // ---- streaming softmax (16 lanes, one row each)
        if (tid < 16) {
            int row = tid, grow = q0 + row;
            float mold = mrow[row], mx = mold;
            float p[32];
            #pragma unroll
            for (int c = 0; c < 32; ++c) {
                float sv = (k0g + c <= grow) ? S[row * 32 + c] * 0.03125f
                                             : -INFINITY;
                p[c] = sv;
                mx = fmaxf(mx, sv);
            }
            float corr = (mold == -INFINITY) ? 0.0f : __expf(mold - mx);
            float sum = 0.0f;
            #pragma unroll
            for (int c = 0; c < 32; ++c) {
                float e = __expf(p[c] - mx);   // masked -> exp(-inf)=0
                sum += e;
                P[row * 32 + c] = f2bf(e);
            }
            lrow[row] = lrow[row] * corr + sum;
            mrow[row] = mx;
            crow[row] = corr;
        }
        __syncthreads();

        // ---- rescale accumulators, then acc += P * V_block
        #pragma unroll
        for (int nt = 0; nt < 4; ++nt)
            #pragma unroll
            for (int r = 0; r < 8; ++r)
                acc[nt][r] *= crow[r + hi8];

        v16bf ap = ldfrag(&P[lm * 32 + hi8]);  // P fragment from LDS
        #pragma unroll
        for (int nt = 0; nt < 4; ++nt) {
            int d = wid * 64 + nt * 16 + lm;   // B column = fixed d
            const unsigned short* vb = Vb + (b * CTX + k0g + hi8) * DIM + d;
            Frag fb;
            #pragma unroll
            for (int k = 0; k < 8; ++k) {      // strided gather of V column
                fb.s[k]     = vb[k * DIM];
                fb.s[8 + k] = vb[(16 + k) * DIM];
            }
            acc[nt] = wmma_bf16(ap, fb.v, acc[nt]);
        }
        __syncthreads();
    }

    // ---- epilogue: /l, + residual h, store hidden as bf16
    #pragma unroll
    for (int nt = 0; nt < 4; ++nt)
        #pragma unroll
        for (int r = 0; r < 8; ++r) {
            int row = r + hi8;
            int idx = (bq + row) * DIM + wid * 64 + nt * 16 + lm;
            float v = acc[nt][r] / lrow[row] + hF[idx];
            hidB[idx] = f2bf(v);
        }
}

// ---------------------------------------------------------------------------
// Kernel 5: logits = hidden @ Wo^T + bo   (M=8192, N=32000, K=1024), f32 out.
// Block = 8 waves -> 128x256 tile (each wave 64x64, 16 accumulators).
// A and B K-chunks (K=64) staged in LDS via async global->LDS loads;
// rows padded to 72 shorts so fragment reads spread across all banks.
// ---------------------------------------------------------------------------
__global__ __launch_bounds__(256)
void k_gemm_logits(const unsigned short* __restrict__ hidB,
                   const unsigned short* __restrict__ WoB,
                   const float* __restrict__ bo,
                   float* __restrict__ out) {
    __shared__ u32x4 ldsA[128 * 9];   // 128 rows x (64 K + 8 pad) bf16 = 18 KB
    __shared__ u32x4 ldsB[256 * 9];   // 256 rows x (64 K + 8 pad) bf16 = 36 KB

    int tid = threadIdx.x, wid = tid >> 5, lane = tid & 31;
    int hi8 = (lane >> 4) * 8, lm = lane & 15;
    int wm = wid >> 2;                // 0..1 : 64-row slice
    int wn = wid & 3;                 // 0..3 : 64-col slice
    int mBase = blockIdx.y * 128;     // 64 m-blocks
    int nBase = blockIdx.x * 256;     // 125 n-blocks (125*256 == 32000)

    const unsigned short* As = (const unsigned short*)ldsA;
    const unsigned short* Bs = (const unsigned short*)ldsB;

    v8f z = {};
    v8f acc[4][4];
    #pragma unroll
    for (int mi = 0; mi < 4; ++mi)
        #pragma unroll
        for (int ni = 0; ni < 4; ++ni) acc[mi][ni] = z;

    for (int k0 = 0; k0 < DIM; k0 += 64) {
        // ---- stage A (128x64) : 1024 x 16B chunks, 4 per thread
        #pragma unroll
        for (int j = 0; j < 4; ++j) {
            int idx = tid + j * 256;
            int row = idx >> 3, c = idx & 7;
            async_ld16(&ldsA[row * 9 + c],
                       hidB + (mBase + row) * DIM + k0 + c * 8);
        }
        // ---- stage B (256x64) : 2048 x 16B chunks, 8 per thread
        #pragma unroll
        for (int j = 0; j < 8; ++j) {
            int idx = tid + j * 256;
            int row = idx >> 3, c = idx & 7;
            async_ld16(&ldsB[row * 9 + c],
                       WoB + (nBase + row) * DIM + k0 + c * 8);
        }
        wait_async0();
        __syncthreads();

        #pragma unroll
        for (int ks = 0; ks < 2; ++ks) {       // two K=32 steps per chunk
            v16bf af[4], bf[4];
            #pragma unroll
            for (int mi = 0; mi < 4; ++mi)
                af[mi] = ldfrag(As + (wm * 64 + mi * 16 + lm) * 72 + ks * 32 + hi8);
            #pragma unroll
            for (int ni = 0; ni < 4; ++ni)
                bf[ni] = ldfrag(Bs + (wn * 64 + ni * 16 + lm) * 72 + ks * 32 + hi8);
            #pragma unroll
            for (int mi = 0; mi < 4; ++mi)
                #pragma unroll
                for (int ni = 0; ni < 4; ++ni)
                    acc[mi][ni] = wmma_bf16(af[mi], bf[ni], acc[mi][ni]);
        }
        __syncthreads();
    }

    #pragma unroll
    for (int ni = 0; ni < 4; ++ni) {
        int   col  = nBase + wn * 64 + ni * 16 + lm;
        float bias = bo[col];
        #pragma unroll
        for (int mi = 0; mi < 4; ++mi)
            #pragma unroll
            for (int r = 0; r < 8; ++r) {
                long row = mBase + wm * 64 + mi * 16 + r + hi8;
                out[row * (long)VOCAB + col] = acc[mi][ni][r] + bias;
            }
    }
}

// ---------------------------------------------------------------------------
// Launch
// ---------------------------------------------------------------------------
extern "C" void kernel_launch(void* const* d_in, const int* in_sizes, int n_in,
                              void* d_out, int out_size, void* d_ws, size_t ws_size,
                              hipStream_t stream) {
    (void)in_sizes; (void)n_in; (void)out_size; (void)ws_size;

    const int*   x     = (const int*)  d_in[0];
    const float* embed = (const float*)d_in[1];
    const float* pe    = (const float*)d_in[2];
    const float* Wq    = (const float*)d_in[3];
    const float* Wk    = (const float*)d_in[4];
    const float* Wv    = (const float*)d_in[5];
    const float* Wo    = (const float*)d_in[6];
    const float* bo    = (const float*)d_in[7];
    float*       out   = (float*)d_out;

    // Workspace layout (bytes)
    char* ws = (char*)d_ws;
    float*          hF   = (float*)(ws);                            // 33.55 MB
    unsigned short* hB   = (unsigned short*)(ws + 33554432);        // 16.78 MB
    unsigned short* Qb   = (unsigned short*)(ws + 50331648);
    unsigned short* Kb   = (unsigned short*)(ws + 67108864);
    unsigned short* Vb   = (unsigned short*)(ws + 83886080);
    unsigned short* hidB = (unsigned short*)(ws + 100663296);
    unsigned short* WqB  = (unsigned short*)(ws + 117440512);
    unsigned short* WkB  = (unsigned short*)(ws + 119537664);
    unsigned short* WvB  = (unsigned short*)(ws + 121634816);
    unsigned short* WoB  = (unsigned short*)(ws + 123731968);       // ends ~189 MB

    // 1) weight downconverts
    k_cvt_bf16<<<(DIM * DIM + 255) / 256, 256, 0, stream>>>(Wq, WqB, DIM * DIM);
    k_cvt_bf16<<<(DIM * DIM + 255) / 256, 256, 0, stream>>>(Wk, WkB, DIM * DIM);
    k_cvt_bf16<<<(DIM * DIM + 255) / 256, 256, 0, stream>>>(Wv, WvB, DIM * DIM);
    k_cvt_bf16<<<(VOCAB * DIM + 255) / 256, 256, 0, stream>>>(Wo, WoB, VOCAB * DIM);

    // 2) embedding + PE
    k_embed<<<(MROWS * DIM) / 256, 256, 0, stream>>>(x, embed, pe, hF, hB);

    // 3) Q/K/V projections (z selects which)
    dim3 gq(512, 1, 3);
    k_gemm_qkv<<<gq, 256, 0, stream>>>(hB, WqB, WkB, WvB, Qb, Kb, Vb);

    // 4) causal attention + residual
    dim3 ga(CTX / 16, BATCH, 1);
    k_attn<<<ga, 512, 0, stream>>>(Qb, Kb, Vb, hF, hidB);

    // 5) logits: 64 x 125 blocks of 128x256
    dim3 gl(125, 64, 1);
    k_gemm_logits<<<gl, 256, 0, stream>>>(hidB, WoB, bo, out);
}
